// SelfAttention_1357209665746
// MI455X (gfx1250) — compile-verified, matrix operand order checked by reading, not run on hardware
//
#include <hip/hip_runtime.h>
#include <hip/hip_bf16.h>

// Problem constants
#define Bc 4
#define Tc 2048
#define Dc 1024
#define BT (Bc * Tc)          // 8192 rows total

typedef __attribute__((ext_vector_type(16))) __bf16 v16bf;
typedef __attribute__((ext_vector_type(8)))  __bf16 v8bf;
typedef __attribute__((ext_vector_type(8)))  float  v8f;

// Load 16 bf16 for one lane of a WMMA A/B operand when the K-dim is
// contiguous in memory. p points at (row_base + k0 + (lane>>4)*8).
// Elements 0..7 <- p[0..7] (K group 0), 8..15 <- p[16..23] (K group 1).
__device__ __forceinline__ v16bf ld16(const __bf16* p) {
    v8bf lo = *(const v8bf*)(p);
    v8bf hi = *(const v8bf*)(p + 16);
    v16bf r;
#pragma unroll
    for (int i = 0; i < 8; ++i) { r[i] = lo[i]; r[i + 8] = hi[i]; }
    return r;
}

__device__ __forceinline__ v8f wmma_bf16(v16bf a, v16bf b, v8f c) {
    return __builtin_amdgcn_wmma_f32_16x16x32_bf16(
        /*neg_a=*/false, a, /*neg_b=*/false, b,
        /*c_mod=*/(short)0, c, /*reuse_a=*/false, /*reuse_b=*/false);
}

// ---------------------------------------------------------------------------
// 0) fp32 -> bf16 conversion
// ---------------------------------------------------------------------------
__global__ void __launch_bounds__(256) cvt_bf16(const float* __restrict__ src,
                                                __bf16* __restrict__ dst, int n) {
    int i = blockIdx.x * 256 + threadIdx.x;
    if (i < n) dst[i] = (__bf16)src[i];
}

// ---------------------------------------------------------------------------
// 1) QKV projection: Out[t][d] = sum_c X[t][c] * W[d][c]   (B = W^T)
//    Wave computes a 32x64 strip: 2 A-tiles x 4 B-tiles, 8 accumulators.
//    mat 0 -> Q (scaled by 1/sqrt(D)), 1 -> K, 2 -> V stored transposed.
// ---------------------------------------------------------------------------
__global__ void __launch_bounds__(256) qkv_gemm(const __bf16* __restrict__ Xb,
                                                const __bf16* __restrict__ Wb,
                                                __bf16* __restrict__ Qb,
                                                __bf16* __restrict__ Kb,
                                                __bf16* __restrict__ Vt) {
    int wave = (blockIdx.x * 256 + threadIdx.x) >> 5;   // 256*48 waves
    int lane = threadIdx.x & 31;
    int mt = wave / 48;              // 0..255 : 32-row tile of the 8192 rows
    int ng = wave % 48;              // 48 groups of 64 output columns (3 mats)
    int mat = ng >> 4;               // 0=Q 1=K 2=V
    int n0  = (ng & 15) << 6;        // column base within D
    int h  = lane >> 4;
    int mn = lane & 15;

    const __bf16* A0 = Xb + (size_t)(mt * 32 +  0 + mn) * Dc + h * 8;
    const __bf16* A1 = Xb + (size_t)(mt * 32 + 16 + mn) * Dc + h * 8;
    const __bf16* Wmat = Wb + (size_t)mat * Dc * Dc;
    const __bf16* B0 = Wmat + (size_t)(n0 +  0 + mn) * Dc + h * 8;
    const __bf16* B1 = Wmat + (size_t)(n0 + 16 + mn) * Dc + h * 8;
    const __bf16* B2 = Wmat + (size_t)(n0 + 32 + mn) * Dc + h * 8;
    const __bf16* B3 = Wmat + (size_t)(n0 + 48 + mn) * Dc + h * 8;

    v8f c00 = {}, c01 = {}, c02 = {}, c03 = {};
    v8f c10 = {}, c11 = {}, c12 = {}, c13 = {};
    for (int k0 = 0; k0 < Dc; k0 += 32) {
        v16bf a0 = ld16(A0 + k0);
        v16bf a1 = ld16(A1 + k0);
        v16bf b0 = ld16(B0 + k0);
        v16bf b1 = ld16(B1 + k0);
        v16bf b2 = ld16(B2 + k0);
        v16bf b3 = ld16(B3 + k0);
        c00 = wmma_bf16(a0, b0, c00);  c10 = wmma_bf16(a1, b0, c10);
        c01 = wmma_bf16(a0, b1, c01);  c11 = wmma_bf16(a1, b1, c11);
        c02 = wmma_bf16(a0, b2, c02);  c12 = wmma_bf16(a1, b2, c12);
        c03 = wmma_bf16(a0, b3, c03);  c13 = wmma_bf16(a1, b3, c13);
    }

    const float scale = (mat == 0) ? 0.03125f : 1.0f;   // 1/sqrt(1024)
    if (mat == 2) {
        // Vt[d][global_t], d-major so the P*V GEMM reads contiguous K
#pragma unroll
        for (int r = 0; r < 8; ++r) {
            int m = r + h * 8;
            size_t g0 = (size_t)(mt * 32 + m);
            size_t g1 = (size_t)(mt * 32 + 16 + m);
            Vt[(size_t)(n0 +  0 + mn) * BT + g0] = (__bf16)c00[r];
            Vt[(size_t)(n0 + 16 + mn) * BT + g0] = (__bf16)c01[r];
            Vt[(size_t)(n0 + 32 + mn) * BT + g0] = (__bf16)c02[r];
            Vt[(size_t)(n0 + 48 + mn) * BT + g0] = (__bf16)c03[r];
            Vt[(size_t)(n0 +  0 + mn) * BT + g1] = (__bf16)c10[r];
            Vt[(size_t)(n0 + 16 + mn) * BT + g1] = (__bf16)c11[r];
            Vt[(size_t)(n0 + 32 + mn) * BT + g1] = (__bf16)c12[r];
            Vt[(size_t)(n0 + 48 + mn) * BT + g1] = (__bf16)c13[r];
        }
    } else {
        __bf16* Ob = (mat == 0) ? Qb : Kb;
#pragma unroll
        for (int r = 0; r < 8; ++r) {
            int m = r + h * 8;
            __bf16* row0 = Ob + (size_t)(mt * 32 + m) * Dc + n0 + mn;
            __bf16* row1 = Ob + (size_t)(mt * 32 + 16 + m) * Dc + n0 + mn;
            row0[ 0] = (__bf16)(c00[r] * scale);
            row0[16] = (__bf16)(c01[r] * scale);
            row0[32] = (__bf16)(c02[r] * scale);
            row0[48] = (__bf16)(c03[r] * scale);
            row1[ 0] = (__bf16)(c10[r] * scale);
            row1[16] = (__bf16)(c11[r] * scale);
            row1[32] = (__bf16)(c12[r] * scale);
            row1[48] = (__bf16)(c13[r] * scale);
        }
    }
}

// ---------------------------------------------------------------------------
// 2) Scores: S[b][q][k] = sum_d Q[b][q][d] * K[b][k][d]  (Q pre-scaled)
//    Wave computes a 16x64 strip (4 kt-tiles share one A load). Groups fully
//    above the diagonal are skipped; partial diagonal groups compute all 4
//    sub-tiles (entries with k > q are masked by softmax, never read).
// ---------------------------------------------------------------------------
__global__ void __launch_bounds__(256) scores_kernel(const __bf16* __restrict__ Qb,
                                                     const __bf16* __restrict__ Kb,
                                                     float* __restrict__ S) {
    int wave = (blockIdx.x * 256 + threadIdx.x) >> 5;   // 4*128*32 waves
    int lane = threadIdx.x & 31;
    int b  = wave >> 12;          // /(128*32)
    int r  = wave & 4095;
    int qt = r >> 5;              // 0..127
    int kg = r & 31;              // group of 4 kt tiles
    if (kg * 4 > qt) return;      // whole group above diagonal
    int h  = lane >> 4;
    int mn = lane & 15;

    const __bf16* Arow = Qb + ((size_t)b * Tc + qt * 16 + mn) * Dc + h * 8;
    const __bf16* B0 = Kb + ((size_t)b * Tc + (kg * 4 + 0) * 16 + mn) * Dc + h * 8;
    const __bf16* B1 = Kb + ((size_t)b * Tc + (kg * 4 + 1) * 16 + mn) * Dc + h * 8;
    const __bf16* B2 = Kb + ((size_t)b * Tc + (kg * 4 + 2) * 16 + mn) * Dc + h * 8;
    const __bf16* B3 = Kb + ((size_t)b * Tc + (kg * 4 + 3) * 16 + mn) * Dc + h * 8;

    v8f c0 = {}, c1 = {}, c2 = {}, c3 = {};
    for (int d0 = 0; d0 < Dc; d0 += 32) {
        v16bf a = ld16(Arow + d0);
        c0 = wmma_bf16(a, ld16(B0 + d0), c0);
        c1 = wmma_bf16(a, ld16(B1 + d0), c1);
        c2 = wmma_bf16(a, ld16(B2 + d0), c2);
        c3 = wmma_bf16(a, ld16(B3 + d0), c3);
    }

    float* Srow = S + ((size_t)b * Tc + qt * 16) * Tc + kg * 64;
#pragma unroll
    for (int rI = 0; rI < 8; ++rI) {
        int m = rI + h * 8;
        float* p = Srow + (size_t)m * Tc + mn;
        p[ 0] = c0[rI];
        p[16] = c1[rI];
        p[32] = c2[rI];
        p[48] = c3[rI];
    }
}

// ---------------------------------------------------------------------------
// 3) Row softmax with causal mask; writes bf16 P in place over the f32 S row.
//    All global reads happen before the first barrier -> in-place is safe.
// ---------------------------------------------------------------------------
__global__ void __launch_bounds__(256) softmax_kernel(float* __restrict__ S) {
    __shared__ float red[256];
    int gid = blockIdx.x;                 // 0..8191 (b*Tc + q)
    int q   = gid & (Tc - 1);
    float* Srow = S + (size_t)gid * Tc;
    int t = threadIdx.x;

    float vals[8];
    float m = -3.0e38f;
#pragma unroll
    for (int j = 0; j < 8; ++j) {
        int k = t + j * 256;
        float s = (k <= q) ? Srow[k] : -3.0e38f;
        vals[j] = s;
        if (s > m) m = s;
    }
    red[t] = m; __syncthreads();
    for (int off = 128; off > 0; off >>= 1) {
        if (t < off) { float o = red[t + off]; if (o > red[t]) red[t] = o; }
        __syncthreads();
    }
    float mx = red[0]; __syncthreads();

    float sum = 0.0f;
#pragma unroll
    for (int j = 0; j < 8; ++j) {
        int k = t + j * 256;
        float e = (k <= q) ? __expf(vals[j] - mx) : 0.0f;
        vals[j] = e; sum += e;
    }
    red[t] = sum; __syncthreads();
    for (int off = 128; off > 0; off >>= 1) {
        if (t < off) red[t] += red[t + off];
        __syncthreads();
    }
    float inv = 1.0f / red[0];

    __bf16* Prow = (__bf16*)Srow;         // packed bf16, row stride stays 2*Tc
#pragma unroll
    for (int j = 0; j < 8; ++j) {
        int k = t + j * 256;
        Prow[k] = (__bf16)(vals[j] * inv);
    }
}

// ---------------------------------------------------------------------------
// 4) O[b][q][d] = sum_k P[b][q][k] * V[b][k][d]
//    Wave computes a 32x64 strip: Vt (B) loads shared across both row-tiles.
//    K loop truncated to (qt+1)*32; P is exactly zero past each row's causal
//    extent, so the extra K contributes nothing.
// ---------------------------------------------------------------------------
__global__ void __launch_bounds__(256) out_gemm(const __bf16* __restrict__ P,
                                                const __bf16* __restrict__ Vt,
                                                float* __restrict__ Out) {
    int wave = (blockIdx.x * 256 + threadIdx.x) >> 5;   // 4*64*16 waves
    int lane = threadIdx.x & 31;
    int b  = wave >> 10;          // /(64*16)
    int r  = wave & 1023;
    int qt = r >> 4;              // 0..63 : 32-row tile
    int d0 = (r & 15) << 6;
    int h  = lane >> 4;
    int mn = lane & 15;

    // P rows live in the f32 S buffer: bf16 element stride 1, row stride 2*Tc
    const __bf16* A0 = P + ((size_t)(b * Tc + qt * 32 + mn)) * (2 * Tc) + h * 8;
    const __bf16* A1 = A0 + (size_t)16 * (2 * Tc);
    const __bf16* B0 = Vt + (size_t)(d0 +  0 + mn) * BT + (size_t)b * Tc + h * 8;
    const __bf16* B1 = Vt + (size_t)(d0 + 16 + mn) * BT + (size_t)b * Tc + h * 8;
    const __bf16* B2 = Vt + (size_t)(d0 + 32 + mn) * BT + (size_t)b * Tc + h * 8;
    const __bf16* B3 = Vt + (size_t)(d0 + 48 + mn) * BT + (size_t)b * Tc + h * 8;

    int kEnd = (qt + 1) * 32;

    v8f c00 = {}, c01 = {}, c02 = {}, c03 = {};
    v8f c10 = {}, c11 = {}, c12 = {}, c13 = {};
    for (int k0 = 0; k0 < kEnd; k0 += 32) {
        v16bf a0 = ld16(A0 + k0);
        v16bf a1 = ld16(A1 + k0);
        v16bf b0 = ld16(B0 + k0);
        v16bf b1 = ld16(B1 + k0);
        v16bf b2 = ld16(B2 + k0);
        v16bf b3 = ld16(B3 + k0);
        c00 = wmma_bf16(a0, b0, c00);  c10 = wmma_bf16(a1, b0, c10);
        c01 = wmma_bf16(a0, b1, c01);  c11 = wmma_bf16(a1, b1, c11);
        c02 = wmma_bf16(a0, b2, c02);  c12 = wmma_bf16(a1, b2, c12);
        c03 = wmma_bf16(a0, b3, c03);  c13 = wmma_bf16(a1, b3, c13);
    }

    float* Orow = Out + ((size_t)(b * Tc + qt * 32)) * Dc + d0 + mn;
#pragma unroll
    for (int rI = 0; rI < 8; ++rI) {
        int m = rI + h * 8;
        float* p0 = Orow + (size_t)m * Dc;
        float* p1 = Orow + (size_t)(m + 16) * Dc;
        p0[ 0] = c00[rI];
        p0[16] = c01[rI];
        p0[32] = c02[rI];
        p0[48] = c03[rI];
        p1[ 0] = c10[rI];
        p1[16] = c11[rI];
        p1[32] = c12[rI];
        p1[48] = c13[rI];
    }
}

// ---------------------------------------------------------------------------
// Launch
// ---------------------------------------------------------------------------
extern "C" void kernel_launch(void* const* d_in, const int* in_sizes, int n_in,
                              void* d_out, int out_size, void* d_ws, size_t ws_size,
                              hipStream_t stream) {
    const float* X  = (const float*)d_in[0];
    const float* Wq = (const float*)d_in[1];
    const float* Wk = (const float*)d_in[2];
    const float* Wv = (const float*)d_in[3];
    float* Out = (float*)d_out;

    char* ws = (char*)d_ws;
    // Workspace layout (bytes):
    //   Xb : 16 MiB  bf16 [8192][1024]
    //   Wb :  6 MiB  bf16 [3][1024][1024]
    //   Qb : 16 MiB  bf16 [8192][1024]   (pre-scaled by 1/sqrt(D))
    //   Kb : 16 MiB  bf16 [8192][1024]
    //   Vt : 16 MiB  bf16 [1024][8192]   (transposed)
    //   S  : 64 MiB  f32  [4][2048][2048] (P bf16 written in place)
    __bf16* Xb = (__bf16*)(ws);
    __bf16* Wb = (__bf16*)(ws + 16777216);
    __bf16* Qb = (__bf16*)(ws + 23068672);
    __bf16* Kb = (__bf16*)(ws + 39845888);
    __bf16* Vt = (__bf16*)(ws + 56623104);
    float*  S  = (float*) (ws + 73400320);

    // 0) convert inputs to bf16
    cvt_bf16<<<32768, 256, 0, stream>>>(X,  Xb, Bc * Tc * Dc);
    cvt_bf16<<<4096,  256, 0, stream>>>(Wq, Wb,               Dc * Dc);
    cvt_bf16<<<4096,  256, 0, stream>>>(Wk, Wb + 1 * Dc * Dc, Dc * Dc);
    cvt_bf16<<<4096,  256, 0, stream>>>(Wv, Wb + 2 * Dc * Dc, Dc * Dc);

    // 1) Q,K,V projections: 256*48 waves = 12288 -> 1536 blocks of 8 waves
    qkv_gemm<<<1536, 256, 0, stream>>>(Xb, Wb, Qb, Kb, Vt);

    // 2) scores: 4*128*32 waves = 16384 -> 2048 blocks
    scores_kernel<<<2048, 256, 0, stream>>>(Qb, Kb, S);

    // 3) softmax: one block per row, 8192 rows
    softmax_kernel<<<8192, 256, 0, stream>>>(S);

    // 4) output GEMM: 4*64*16 waves = 4096 -> 512 blocks
    out_gemm<<<512, 256, 0, stream>>>((const __bf16*)S, Vt, Out);
}
